// GatherModel_19344532701718
// MI455X (gfx1250) — matrix operand order repaired and checked by reading, not compiled, not run on hardware
//
#include <hip/hip_runtime.h>
#include <stddef.h>

#define NN 50000
#define EE 150000
#define DIN 42
#define DH 42
#define EIN 10
#define EH 128
#define STEPS 6
#define NOUT (DH*DH)     // 1764
#define NPAD 1776        // 111 * 16
#define NT 111           // N tiles
#define MT (EE/16)       // 9375 M tiles

typedef _Float16 v16h __attribute__((ext_vector_type(16)));
typedef _Float16 v8h  __attribute__((ext_vector_type(8)));
typedef _Float16 h2   __attribute__((ext_vector_type(2)));
typedef float    v8f  __attribute__((ext_vector_type(8)));

// -------- out = relu(n_feat @ lin0_w + lin0_b) --------
__global__ void k_lin0(const float* __restrict__ nf, const float* __restrict__ w,
                       const float* __restrict__ b, float* __restrict__ out) {
  __shared__ float sw[DIN*DH];
  __shared__ float sb[DH];
  __shared__ float sx[6][DIN];
  int t = threadIdx.x;
  for (int i = t; i < DIN*DH; i += 256) sw[i] = w[i];
  if (t < DH) sb[t] = b[t];
  int node0 = blockIdx.x * 6;
  for (int i = t; i < 6*DIN; i += 256) {
    int nl = i / DIN, ii = i % DIN;
    int node = node0 + nl;
    sx[nl][ii] = (node < NN) ? nf[node*DIN + ii] : 0.f;
  }
  __syncthreads();
  if (t < 6*DH) {
    int nl = t / DH, j = t % DH;
    int node = node0 + nl;
    if (node < NN) {
      float acc = sb[j];
      #pragma unroll 6
      for (int i = 0; i < DIN; i++) acc += sx[nl][i] * sw[i*DH + j];
      out[node*DH + j] = fmaxf(acc, 0.f);
    }
  }
}

// -------- h16 = f16(relu(e_feat @ en1_w + en1_b)) --------
__global__ void k_edge1(const float* __restrict__ ef, const float* __restrict__ w,
                        const float* __restrict__ b, _Float16* __restrict__ h16) {
  __shared__ float sw[EIN*EH];
  __shared__ float sb[EH];
  __shared__ float sx[2][EIN];
  int t = threadIdx.x;
  for (int i = t; i < EIN*EH; i += 256) sw[i] = w[i];
  if (t < EH) sb[t] = b[t];
  int e0 = blockIdx.x * 2;
  if (t < 2*EIN) {
    int el = t / EIN, ii = t % EIN;
    int e = e0 + el;
    sx[el][ii] = (e < EE) ? ef[e*EIN + ii] : 0.f;
  }
  __syncthreads();
  int el = t >> 7, j = t & 127;
  int e = e0 + el;
  if (e < EE) {
    float acc = sb[j];
    #pragma unroll
    for (int i = 0; i < EIN; i++) acc += sx[el][i] * sw[i*EH + j];
    h16[e*EH + j] = (_Float16)fmaxf(acc, 0.f);
  }
}

// -------- pack en2_w into WMMA B-fragment order (f16, zero-padded N) --------
// B 16-bit 32x16 layout: lane L holds column N=L&15, K = (L>>4)*8 + (j&7) + (j>>3)*16, j=0..15.
__global__ void k_pack_b(const float* __restrict__ en2w, _Float16* __restrict__ bp) {
  int idx = blockIdx.x * 256 + threadIdx.x;
  const int total = NT * 4 * 32 * 16;
  if (idx >= total) return;
  int j     = idx & 15;
  int lane  = (idx >> 4) & 31;
  int kt    = (idx >> 9) & 3;
  int ntile = idx >> 11;
  int k = kt*32 + ((lane >> 4) << 3) + (j & 7) + ((j >> 3) << 4);
  int n = ntile*16 + (lane & 15);
  float v = (n < NOUT) ? en2w[k*NOUT + n] : 0.f;
  bp[idx] = (_Float16)v;
}

// -------- W_e16 = f16(h16 @ en2_w + en2_b), via v_wmma_f32_16x16x32_f16 --------
// C tile staged through per-wave LDS so global stores are full b128 per lane.
__global__ void k_gemm_we(const _Float16* __restrict__ h16, const _Float16* __restrict__ bp,
                          const float* __restrict__ bias, _Float16* __restrict__ we) {
  __shared__ __attribute__((aligned(32))) _Float16 st[8][16][16]; // [wave][row][col]
  int wv   = threadIdx.x >> 5;
  int wid  = (int)((blockIdx.x * 256 + threadIdx.x) >> 5);
  int lane = threadIdx.x & 31;
  if (wid >= MT*NT) return;
  int mtile = wid / NT;
  int ntile = wid % NT;
  int half_sel = lane >> 4;
  int lrow     = lane & 15;
  // A 16-bit 16x32 layout: lanes<16 hold K {0..7,16..23}; lanes>=16 hold K {8..15,24..31}
  const _Float16* aptr = h16 + (size_t)(mtile*16 + lrow) * EH + half_sel*8;
  const _Float16* bptr = bp + (size_t)ntile * 2048 + lane*16;
  v8f c = {};
  #pragma unroll
  for (int kt = 0; kt < 4; kt++) {
    v8h alo = *(const v8h*)(aptr + kt*32);
    v8h ahi = *(const v8h*)(aptr + kt*32 + 16);
    v16h a = __builtin_shufflevector(alo, ahi, 0,1,2,3,4,5,6,7,8,9,10,11,12,13,14,15);
    v16h bf = *(const v16h*)(bptr + kt*512);
    c = __builtin_amdgcn_wmma_f32_16x16x32_f16(false, a, false, bf,
                                               (short)0, c, false, false);
  }
  // C layout: VGPR r, lane L -> M = (L>=16?8:0)+r, N = L&15
  int ncol = ntile*16 + lrow;
  float bc = (ncol < NOUT) ? bias[ncol] : 0.f;
  #pragma unroll
  for (int r = 0; r < 8; r++) {
    st[wv][half_sel*8 + r][lrow] = (_Float16)(c[r] + bc);
  }
  // intra-wave LDS transpose: wait for the ds stores, no workgroup barrier needed
  asm volatile("s_wait_dscnt 0x0" ::: "memory");
  int row = lane >> 1, hh = lane & 1;
  v8h vals = *(const v8h*)&st[wv][row][hh*8];
  size_t dstoff = (size_t)(mtile*16 + row) * NPAD + (size_t)(ntile*16 + hh*8);
  *(v8h*)(we + dstoff) = vals;   // aligned 16B store, full coalescing
}

// -------- msg = out[src] . W_e ; atomic scatter-sum into aggr[dst] --------
// x row hoisted into registers (fully unrolled), inner loop = 1 load + 2 fma.
__global__ void k_msg(const float* __restrict__ out, const _Float16* __restrict__ we,
                      const int* __restrict__ src, const int* __restrict__ dst,
                      float* __restrict__ aggr) {
  int wave = threadIdx.x >> 5;
  int lane = threadIdx.x & 31;
  int e = blockIdx.x * 8 + wave;
  if (e >= EE) return;
  int s = src[e], d = dst[e];
  if (lane >= 21) return;
  const float2* xr2 = (const float2*)(out + (size_t)s * DH);  // 8B aligned (42*4)
  float x[DH];
  #pragma unroll
  for (int i = 0; i < DH/2; i++) {
    float2 v = xr2[i];
    x[2*i] = v.x; x[2*i+1] = v.y;
  }
  const _Float16* wrow = we + (size_t)e * NPAD + 2*lane;
  float m0 = 0.f, m1 = 0.f;
  #pragma unroll
  for (int i = 0; i < DH; i++) {
    h2 wv = *(const h2*)(wrow + i*DH);    // contiguous 84B per (e,i) across lanes
    m0 += x[i] * (float)wv[0];
    m1 += x[i] * (float)wv[1];
  }
  atomicAdd(&aggr[(size_t)d*DH + 2*lane],     m0);
  atomicAdd(&aggr[(size_t)d*DH + 2*lane + 1], m1);
}

// -------- m = relu(aggr+out+cb); out = [m,out] @ msg_w + msg_b ; final: +init --------
__global__ void k_update(const float* __restrict__ aggr, float* __restrict__ out,
                         const float* __restrict__ cb, const float* __restrict__ mw,
                         const float* __restrict__ mb, const float* __restrict__ nf,
                         float* __restrict__ dout, int final_step) {
  __shared__ float sw[2*DH*DH];
  __shared__ float sb[DH];
  __shared__ float sm[6][DH];
  __shared__ float so[6][DH];
  int t = threadIdx.x;
  for (int i = t; i < 2*DH*DH; i += 256) sw[i] = mw[i];
  if (t < DH) sb[t] = mb[t];
  int node0 = blockIdx.x * 6;
  for (int i = t; i < 6*DH; i += 256) {
    int nl = i / DH, k = i % DH;
    int node = node0 + nl;
    if (node < NN) {
      float o = out[node*DH + k];
      sm[nl][k] = fmaxf(aggr[node*DH + k] + o + cb[k], 0.f);
      so[nl][k] = o;
    }
  }
  __syncthreads();
  if (t < 6*DH) {
    int nl = t / DH, j = t % DH;
    int node = node0 + nl;
    if (node < NN) {
      float acc = sb[j];
      #pragma unroll 6
      for (int k = 0; k < DH; k++) {
        acc += sm[nl][k] * sw[k*DH + j];
        acc += so[nl][k] * sw[(DH + k)*DH + j];
      }
      if (final_step) dout[node*DH + j] = acc + nf[node*DH + j];
      else            out[node*DH + j]  = acc;
    }
  }
}

static inline size_t align256(size_t x) { return (x + 255) & ~(size_t)255; }

extern "C" void kernel_launch(void* const* d_in, const int* in_sizes, int n_in,
                              void* d_out, int out_size, void* d_ws, size_t ws_size,
                              hipStream_t stream) {
  (void)in_sizes; (void)n_in; (void)out_size; (void)ws_size;
  const float* nf  = (const float*)d_in[0];
  const float* ef  = (const float*)d_in[1];
  const int*   src = (const int*)d_in[2];
  const int*   dst = (const int*)d_in[3];
  const float* l0w = (const float*)d_in[4];
  const float* l0b = (const float*)d_in[5];
  const float* e1w = (const float*)d_in[6];
  const float* e1b = (const float*)d_in[7];
  const float* e2w = (const float*)d_in[8];
  const float* e2b = (const float*)d_in[9];
  const float* cb  = (const float*)d_in[10];
  const float* mw  = (const float*)d_in[11];
  const float* mb  = (const float*)d_in[12];
  float* dout = (float*)d_out;

  char* ws = (char*)d_ws;
  size_t off = 0;
  float* out  = (float*)(ws + off); off += align256((size_t)NN * DH * sizeof(float));
  float* aggr = (float*)(ws + off); off += align256((size_t)NN * DH * sizeof(float));
  _Float16* h16 = (_Float16*)(ws + off); off += align256((size_t)EE * EH * sizeof(_Float16));
  _Float16* bp  = (_Float16*)(ws + off); off += align256((size_t)NT * 2048 * sizeof(_Float16));
  _Float16* we  = (_Float16*)(ws + off); off += align256((size_t)EE * NPAD * sizeof(_Float16));

  // One-time (per call) precompute
  k_lin0  <<<(NN + 5) / 6, 256, 0, stream>>>(nf, l0w, l0b, out);
  k_edge1 <<<(EE + 1) / 2, 256, 0, stream>>>(ef, e1w, e1b, h16);
  k_pack_b<<<(NT*2048 + 255) / 256, 256, 0, stream>>>(e2w, bp);
  {
    long long waves = (long long)MT * NT;
    int blocks = (int)((waves + 7) / 8);
    k_gemm_we<<<blocks, 256, 0, stream>>>(h16, bp, e2b, we);
  }

  // 6 message-passing steps
  for (int step = 0; step < STEPS; step++) {
    hipMemsetAsync(aggr, 0, (size_t)NN * DH * sizeof(float), stream);
    k_msg   <<<(EE + 7) / 8, 256, 0, stream>>>(out, we, src, dst, aggr);
    k_update<<<(NN + 5) / 6, 256, 0, stream>>>(aggr, out, cb, mw, mb, nf, dout,
                                               step == STEPS - 1 ? 1 : 0);
  }
}
